// ContentMultiheadAttention_523986010170
// MI455X (gfx1250) — matrix operand order repaired and checked by reading, not compiled, not run on hardware
//
#include <hip/hip_runtime.h>
#include <math.h>

// ---------------------------------------------------------------------------
// MultiheadAttention, S=2048 B=4 E=512 H=8 D=64, fp32 in/out.
// Bandwidth-bound (~130MB compulsory vs 52 GFLOP):
//  - bf16 WMMA (v_wmma_f32_16x16x32_bf16), f32 accumulate
//  - weights pre-converted to bf16; GEMM blocks stage their 64KB weight slab
//    into LDS via global_load_async_to_lds_b128, inner loop = ds_load + wmma
//  - flash attention: 8 waves/block share one (b,h); K/V tiles async-staged
//    into LDS (double-buffered); score matrix never touches HBM
// ---------------------------------------------------------------------------

typedef __bf16 bhalf_t;
typedef __attribute__((ext_vector_type(16))) __bf16 v16bf;
typedef __attribute__((ext_vector_type(4)))  __bf16 v4bf;
typedef __attribute__((ext_vector_type(8)))  float  v8f;
typedef __attribute__((ext_vector_type(4)))  int    i4;

#define S_LEN 2048
#define BATCH 4
#define EMB   512
#define HEADS 8
#define HDIM  64

__device__ __forceinline__ v8f wmma_bf16(v16bf a, v16bf b, v8f c) {
  return __builtin_amdgcn_wmma_f32_16x16x32_bf16(false, a, false, b,
                                                 (short)0, c, false, false);
}

// A-fragment (16x32 bf16): lane = row m (lane%16); K chunks at
// [half*8..+7] and [16+half*8..+7] -> two contiguous 16B loads.
__device__ __forceinline__ v16bf load_a_frag(const bhalf_t* lo, const bhalf_t* hi) {
  v16bf a;
  ((i4*)&a)[0] = *(const i4*)lo;
  ((i4*)&a)[1] = *(const i4*)(hi);
  return a;
}

// B-fragment (32x16 bf16): lane = col n (lane%16); k chunks at
// [half*8..+7] and [16+half*8..+7] -> two 16B loads.
__device__ __forceinline__ v16bf load_b_frag(const bhalf_t* p) {
  v16bf b;
  ((i4*)&b)[0] = *(const i4*)p;
  ((i4*)&b)[1] = *(const i4*)(p + 16);
  return b;
}

__device__ __forceinline__ v16bf cvt_a_frag(const float* lo, const float* hi) {
  v16bf a;
#pragma unroll
  for (int i = 0; i < 8; ++i) a[i]     = (bhalf_t)lo[i];
#pragma unroll
  for (int i = 0; i < 8; ++i) a[8 + i] = (bhalf_t)hi[i];
  return a;
}

// ---- CDNA5 async copy to LDS --------------------------------------------
__device__ __forceinline__ unsigned lds_off_u32(const void* p) {
  return (unsigned)(unsigned long long)
      (__attribute__((address_space(3))) const char*)p;
}

__device__ __forceinline__ void async_ld16(unsigned loff, const bhalf_t* g) {
  asm volatile("global_load_async_to_lds_b128 %0, %1, off"
               :: "v"(loff), "v"((unsigned long long)(size_t)g)
               : "memory");
}

// ---------------------------------------------------------------------------
// Kernel 0: f32 -> bf16 conversion (weights).
// ---------------------------------------------------------------------------
__global__ __launch_bounds__(256) void cvt_bf16_kernel(
    const float* __restrict__ src, bhalf_t* __restrict__ dst, int n4) {
  const int i = blockIdx.x * blockDim.x + threadIdx.x;
  if (i < n4) {
    const float4 v = ((const float4*)src)[i];
    v4bf o;
    o[0] = (bhalf_t)v.x; o[1] = (bhalf_t)v.y;
    o[2] = (bhalf_t)v.z; o[3] = (bhalf_t)v.w;
    ((v4bf*)dst)[i] = o;
  }
}

// ---------------------------------------------------------------------------
// Kernel 1: fused QKV projection.  X[S*B,512](f32) @ Wbf^T -> bf16:
//   Q: [B,H,S,D] (pre-scaled 1/sqrt(D)),  K: [B,H,S,D],  V: [B,H,D,S].
// Block = 8 waves sharing one 64-col weight slab (async-staged to LDS, 64KB);
// each wave computes a 16x64 tile; inner loop: global A (pipelined) + LDS B
// (all 4 fragments loaded before the 4 WMMAs -> partial dscnt waits).
// ---------------------------------------------------------------------------
__global__ __launch_bounds__(256) void qkv_proj_kernel(
    const float* __restrict__ query, const float* __restrict__ key,
    const float* __restrict__ value, const bhalf_t* __restrict__ wbf,
    const float* __restrict__ bias,
    bhalf_t* __restrict__ Qbf, bhalf_t* __restrict__ Kbf,
    bhalf_t* __restrict__ Vt) {
  __shared__ __align__(16) bhalf_t wlds[64 * EMB];   // 64 KB weight slab

  const int tid  = threadIdx.x;
  const int lane = tid & 31;
  const int hl   = lane >> 4;
  const int lrow = lane & 15;

  // 64 blocks per n-tile -> all 8 waves of a block share nt (same weights)
  const int nt    = blockIdx.x >> 6;                 // 0..23
  const int mt    = (blockIdx.x & 63) * 8 + (tid >> 5);
  const int proj  = nt >> 3;                         // 0=q 1=k 2=v
  const int ncol0 = (nt & 7) * 64;

  // ---- stage 64 x 512 bf16 weight slab into LDS (16 x 16B per thread) ----
  const bhalf_t* wbase = wbf + (size_t)(proj * EMB + ncol0) * EMB;
  {
    const int row0 = tid >> 6;          // 0..3
    const int off  = (tid & 63) * 8;    // 16B chunk within a 1KB row
#pragma unroll
    for (int j = 0; j < 16; ++j) {
      const int row = j * 4 + row0;
      async_ld16(lds_off_u32(&wlds[row * EMB + off]),
                 wbase + (size_t)row * EMB + off);
    }
  }
  asm volatile("s_wait_asynccnt 0x0" ::: "memory");
  __syncthreads();

  const float* X = (proj == 0) ? query : (proj == 1 ? key : value);
  const float* xrow = X + (size_t)(mt * 16 + lrow) * EMB;
  const bhalf_t* wl = &wlds[lrow * EMB + hl * 8];    // per-lane slab base

  v8f acc[4];
#pragma unroll
  for (int nb = 0; nb < 4; ++nb) acc[nb] = (v8f)0.0f;

  // pipelined k-loop: 16 chunks of 32; A from global (double-buffered),
  // 4 B fragments batched from LDS, then 4 WMMAs.
  v16bf a_c = cvt_a_frag(xrow + hl * 8, xrow + 16 + hl * 8);
#pragma unroll
  for (int i = 0; i < 16; ++i) {
    const int kc = i * 32;
    v16bf a_n;
    if (i < 15) {
      const int kcn = kc + 32;
      a_n = cvt_a_frag(xrow + kcn + hl * 8, xrow + kcn + 16 + hl * 8);
    }
    v16bf bfr[4];
#pragma unroll
    for (int nb = 0; nb < 4; ++nb)
      bfr[nb] = load_b_frag(wl + nb * 16 * EMB + kc);
#pragma unroll
    for (int nb = 0; nb < 4; ++nb)
      acc[nb] = wmma_bf16(a_c, bfr[nb], acc[nb]);
    if (i < 15) a_c = a_n;
  }

#pragma unroll
  for (int nb = 0; nb < 4; ++nb) {
#pragma unroll
    for (int r = 0; r < 8; ++r) {
      const int m   = r + hl * 8;
      const int row = mt * 16 + m;          // row = s*B + b
      const int s   = row >> 2;
      const int b   = row & 3;
      const int e   = ncol0 + nb * 16 + lrow;
      const float val = acc[nb][r] + bias[proj * EMB + e];
      const int h = e >> 6;
      const int d = e & 63;
      const size_t bh = (size_t)(b * HEADS + h);
      if (proj == 0)
        Qbf[(bh * S_LEN + s) * HDIM + d] = (bhalf_t)(val * 0.125f);
      else if (proj == 1)
        Kbf[(bh * S_LEN + s) * HDIM + d] = (bhalf_t)val;
      else
        Vt[(bh * HDIM + d) * S_LEN + s]  = (bhalf_t)val;
    }
  }
}

// ---------------------------------------------------------------------------
// Kernel 2: flash attention.  Block = 8 waves sharing one (b,h).  Per 32-key
// step the block async-copies the K tile (32x64, 4KB) and V^T tile (64x32,
// 4KB) into LDS (double-buffered); each wave: 4 score WMMAs + online softmax
// + 4 PV WMMAs from LDS.  P re-staged C->A layout via 1KB LDS per wave.
// ---------------------------------------------------------------------------
__global__ __launch_bounds__(256) void flash_attn_kernel(
    const bhalf_t* __restrict__ Qbf, const bhalf_t* __restrict__ Kbf,
    const bhalf_t* __restrict__ Vt, const float* __restrict__ mask,
    bhalf_t* __restrict__ Obf) {
  __shared__ __align__(16) bhalf_t kbuf[2][32 * 64];   // 2 x 4KB
  __shared__ __align__(16) bhalf_t vbuf[2][64 * 32];   // 2 x 4KB
  __shared__ __align__(16) bhalf_t pbuf[8][16 * 32];   // 8 x 1KB

  const int tid   = threadIdx.x;
  const int lane  = tid & 31;
  const int hl    = lane >> 4;
  const int lrow  = lane & 15;
  const int wslot = tid >> 5;

  const int bh    = blockIdx.x >> 4;                // 0..31  (b*H+h)
  const int stile = (blockIdx.x & 15) * 8 + wslot;  // q-tile of this wave
  const int b     = bh >> 3;
  const int h     = bh & 7;

  const bhalf_t* Qb = Qbf + (size_t)bh * S_LEN * HDIM;
  const bhalf_t* Kb = Kbf + (size_t)bh * S_LEN * HDIM;
  const bhalf_t* Vb = Vt  + (size_t)bh * HDIM * S_LEN;
  const float* mrow_base =
      mask + (size_t)b * S_LEN * S_LEN + (size_t)(stile * 16) * S_LEN;

  v16bf qf[2];
  {
    const bhalf_t* qrow = Qb + (size_t)(stile * 16 + lrow) * HDIM;
    qf[0] = load_a_frag(qrow + hl * 8,      qrow + 16 + hl * 8);
    qf[1] = load_a_frag(qrow + 32 + hl * 8, qrow + 48 + hl * 8);
  }

  v8f o[4];
#pragma unroll
  for (int nb = 0; nb < 4; ++nb) o[nb] = (v8f)0.0f;
  float rmax[8], rsum[8];
#pragma unroll
  for (int r = 0; r < 8; ++r) { rmax[r] = -3.0e38f; rsum[r] = 0.0f; }

  bhalf_t* P = pbuf[wslot];

  auto issue_tile = [&](int kt, int buf) {
    {  // K tile: 32 rows x 128B; thread -> row tid/8, 16B chunk tid%8
      const int row = tid >> 3, sub = (tid & 7) * 8;
      async_ld16(lds_off_u32(&kbuf[buf][row * 64 + sub]),
                 Kb + (size_t)(kt + row) * HDIM + sub);
    }
    {  // V tile: 64 rows x 64B; thread -> row tid/4, 16B chunk tid%4
      const int row = tid >> 2, sub = (tid & 3) * 8;
      async_ld16(lds_off_u32(&vbuf[buf][row * 32 + sub]),
                 Vb + (size_t)row * S_LEN + kt + sub);
    }
  };

  issue_tile(0, 0);

  const int NIT = S_LEN / 32;   // 64
  for (int it = 0; it < NIT; ++it) {
    const int kt  = it * 32;
    const int cur = it & 1;
    if (it + 1 < NIT) {
      issue_tile(kt + 32, cur ^ 1);
      asm volatile("s_wait_asynccnt 0x2" ::: "memory");  // current tile done
    } else {
      asm volatile("s_wait_asynccnt 0x0" ::: "memory");
    }
    __syncthreads();

    const bhalf_t* Kt = kbuf[cur];
    const bhalf_t* Vl = vbuf[cur];

    v16bf kb0a = load_b_frag(Kt + lrow * 64 + hl * 8);
    v16bf kb0b = load_b_frag(Kt + lrow * 64 + 32 + hl * 8);
    v16bf kb1a = load_b_frag(Kt + (16 + lrow) * 64 + hl * 8);
    v16bf kb1b = load_b_frag(Kt + (16 + lrow) * 64 + 32 + hl * 8);
    v8f s0 = (v8f)0.0f, s1 = (v8f)0.0f;
    s0 = wmma_bf16(qf[0], kb0a, s0);
    s0 = wmma_bf16(qf[1], kb0b, s0);
    s1 = wmma_bf16(qf[0], kb1a, s1);
    s1 = wmma_bf16(qf[1], kb1b, s1);

    v16bf vb[4];
#pragma unroll
    for (int nb = 0; nb < 4; ++nb)
      vb[nb] = load_b_frag(Vl + (nb * 16 + lrow) * 32 + hl * 8);

    float bm[8];
#pragma unroll
    for (int r = 0; r < 8; ++r) {
      const float* mr = mrow_base + (size_t)(r + hl * 8) * S_LEN + kt;
      const float v0 = s0[r] + mr[lrow];
      const float v1 = s1[r] + mr[16 + lrow];
      s0[r] = v0; s1[r] = v1;
      float t = fmaxf(v0, v1);
      t = fmaxf(t, __shfl_xor(t, 1));
      t = fmaxf(t, __shfl_xor(t, 2));
      t = fmaxf(t, __shfl_xor(t, 4));
      t = fmaxf(t, __shfl_xor(t, 8));
      bm[r] = t;
    }

#pragma unroll
    for (int r = 0; r < 8; ++r) {
      const float nm   = fmaxf(rmax[r], bm[r]);
      const float corr = __expf(rmax[r] - nm);
      const float p0   = __expf(s0[r] - nm);
      const float p1   = __expf(s1[r] - nm);
      float ts = p0 + p1;
      ts += __shfl_xor(ts, 1);
      ts += __shfl_xor(ts, 2);
      ts += __shfl_xor(ts, 4);
      ts += __shfl_xor(ts, 8);
      rsum[r] = rsum[r] * corr + ts;
      rmax[r] = nm;
#pragma unroll
      for (int nb = 0; nb < 4; ++nb) o[nb][r] *= corr;
      const int m = r + hl * 8;
      P[m * 32 + lrow]      = (bhalf_t)p0;
      P[m * 32 + 16 + lrow] = (bhalf_t)p1;
    }
    asm volatile("s_wait_dscnt 0x0" ::: "memory");

    v16bf pa = load_a_frag(P + lrow * 32 + hl * 8,
                           P + lrow * 32 + 16 + hl * 8);
#pragma unroll
    for (int nb = 0; nb < 4; ++nb) o[nb] = wmma_bf16(pa, vb[nb], o[nb]);

    __syncthreads();
  }

#pragma unroll
  for (int nb = 0; nb < 4; ++nb) {
#pragma unroll
    for (int r = 0; r < 8; ++r) {
      const int m = r + hl * 8;
      const int s = stile * 16 + m;
      const int e = h * HDIM + nb * 16 + lrow;
      Obf[((size_t)s * BATCH + b) * EMB + e] = (bhalf_t)(o[nb][r] / rsum[r]);
    }
  }
}

// ---------------------------------------------------------------------------
// Kernel 3: output projection.  Obf[8192,512](bf16) @ Wout_bf^T + bias -> f32.
// Same LDS weight-slab scheme as qkv (8 waves share one 64-col slab).
// ---------------------------------------------------------------------------
__global__ __launch_bounds__(256) void out_proj_kernel(
    const bhalf_t* __restrict__ Obf, const bhalf_t* __restrict__ wbf,
    const float* __restrict__ bias, float* __restrict__ out) {
  __shared__ __align__(16) bhalf_t wlds[64 * EMB];   // 64 KB weight slab

  const int tid  = threadIdx.x;
  const int lane = tid & 31;
  const int hl   = lane >> 4;
  const int lrow = lane & 15;

  const int nt = blockIdx.x >> 6;                    // 0..7
  const int mt = (blockIdx.x & 63) * 8 + (tid >> 5);

  const bhalf_t* wbase = wbf + (size_t)(nt * 64) * EMB;
  {
    const int row0 = tid >> 6;
    const int off  = (tid & 63) * 8;
#pragma unroll
    for (int j = 0; j < 16; ++j) {
      const int row = j * 4 + row0;
      async_ld16(lds_off_u32(&wlds[row * EMB + off]),
                 wbase + (size_t)row * EMB + off);
    }
  }
  asm volatile("s_wait_asynccnt 0x0" ::: "memory");
  __syncthreads();

  const bhalf_t* orow = Obf + (size_t)(mt * 16 + lrow) * EMB;
  const bhalf_t* wl = &wlds[lrow * EMB + hl * 8];

  v8f acc[4];
#pragma unroll
  for (int nb = 0; nb < 4; ++nb) acc[nb] = (v8f)0.0f;

  v16bf a_c = load_a_frag(orow + hl * 8, orow + 16 + hl * 8);
#pragma unroll
  for (int i = 0; i < 16; ++i) {
    const int kc = i * 32;
    v16bf a_n;
    if (i < 15) {
      const int kcn = kc + 32;
      a_n = load_a_frag(orow + kcn + hl * 8, orow + kcn + 16 + hl * 8);
    }
    v16bf bfr[4];
#pragma unroll
    for (int nb = 0; nb < 4; ++nb)
      bfr[nb] = load_b_frag(wl + nb * 16 * EMB + kc);
#pragma unroll
    for (int nb = 0; nb < 4; ++nb)
      acc[nb] = wmma_bf16(a_c, bfr[nb], acc[nb]);
    if (i < 15) a_c = a_n;
  }

#pragma unroll
  for (int nb = 0; nb < 4; ++nb) {
#pragma unroll
    for (int r = 0; r < 8; ++r) {
      const int row = mt * 16 + r + hl * 8;
      const int e   = nt * 64 + nb * 16 + lrow;
      out[(size_t)row * EMB + e] = acc[nb][r] + bias[e];
    }
  }
}

// ---------------------------------------------------------------------------
extern "C" void kernel_launch(void* const* d_in, const int* in_sizes, int n_in,
                              void* d_out, int out_size, void* d_ws,
                              size_t ws_size, hipStream_t stream) {
  const float* query = (const float*)d_in[0];   // [S,B,E]
  const float* key   = (const float*)d_in[1];   // [S,B,E]
  const float* value = (const float*)d_in[2];   // [S,B,E]
  const float* mask  = (const float*)d_in[3];   // [B,S,S]
  const float* in_w  = (const float*)d_in[4];   // [3E,E]
  const float* in_b  = (const float*)d_in[5];   // [3E]
  const float* out_w = (const float*)d_in[6];   // [E,E]
  const float* out_b = (const float*)d_in[7];   // [E]
  float* out = (float*)d_out;

  const size_t n    = (size_t)BATCH * HEADS * S_LEN * HDIM;  // 4,194,304
  const size_t nwin = (size_t)3 * EMB * EMB;                 //   786,432
  const size_t nwo  = (size_t)EMB * EMB;                     //   262,144
  bhalf_t* Qbf   = (bhalf_t*)d_ws;        // 8 MB  [B,H,S,D] (pre-scaled)
  bhalf_t* Kbf   = Qbf + n;               // 8 MB  [B,H,S,D]
  bhalf_t* Vt    = Kbf + n;               // 8 MB  [B,H,D,S]
  bhalf_t* Obf   = Vt + n;                // 8 MB  [S,B,E]
  bhalf_t* Winbf = Obf + n;               // 1.5 MB
  bhalf_t* Wobf  = Winbf + nwin;          // 0.5 MB

  cvt_bf16_kernel<<<(int)(nwin / 4 / 256), 256, 0, stream>>>(in_w, Winbf,
                                                             (int)(nwin / 4));
  cvt_bf16_kernel<<<(int)(nwo / 4 / 256), 256, 0, stream>>>(out_w, Wobf,
                                                            (int)(nwo / 4));
  // 24 n-tiles x 64 block-groups; 8 waves/block (one 16x64 tile each)
  qkv_proj_kernel<<<1536, 256, 0, stream>>>(query, key, value, Winbf, in_b,
                                            Qbf, Kbf, Vt);
  // 32 (b,h) x 16 q-tile groups, 8 waves/block
  flash_attn_kernel<<<512, 256, 0, stream>>>(Qbf, Kbf, Vt, mask, Obf);
  // 8 n-tiles x 64 block-groups, 8 waves/block
  out_proj_kernel<<<512, 256, 0, stream>>>(Obf, Wobf, out_b, out);
}
